// ListNetLoss_33904471835382
// MI455X (gfx1250) — matrix operand order repaired and checked by reading, not compiled, not run on hardware
//
#include <hip/hip_runtime.h>
#include <hip/hip_bf16.h>

typedef __attribute__((ext_vector_type(16))) _Float16 v16h;
typedef __attribute__((ext_vector_type(8)))  float    v8f;

constexpr int B_ = 2048;
constexpr int N_ = 8192;
constexpr int L_ = 512;
constexpr int WAVES = 8;
constexpr int BLOCK = WAVES * 32;
#define NEGBIG (-1e30f)

__device__ __forceinline__ float sanitize_f(float x) {
    // nan -> 0, +inf -> 1e6, -inf -> -1e6, finite unchanged
    if (x != x) return 0.0f;
    if (x == __builtin_inff()) return 1e6f;
    if (x == -__builtin_inff()) return -1e6f;
    return x;
}

// One wave32 per row. Element layout for load/accumulate phases: j = t*32 + lane,
// t = 0..15 (coalesced 128B index loads). Suffix logsumexp via:
//   e[j] = exp(gm[j]-M);  suffsum[j] = Sigma_{k>=j} e[k]
// computed as two 16x16 WMMAs (within-chunk upper-triangular matvec) plus a
// 32-lane shuffle suffix-scan over chunk totals.
__global__ __launch_bounds__(BLOCK) void listnet_rows_kernel(
    const float* __restrict__ scores,   // [B, N]
    const int*   __restrict__ ytrue,    // [B, L]
    float* __restrict__ row_ll,         // [B]
    float* __restrict__ row_used)       // [B]
{
    __shared__ _Float16 eh[WAVES][L_];  // exp(gm - M) staged for WMMA B operand
    __shared__ float    sf[WAVES][L_];  // suffix sums back in j-order

    const int lane = threadIdx.x & 31;
    const int w    = threadIdx.x >> 5;
    const int row  = blockIdx.x * WAVES + w;

    const float* srow = scores + (size_t)row * N_;
    const int*   yrow = ytrue  + (size_t)row * L_;

    // ---- gather + sanitize + mask, track row max ----
    float gm16[16];
    unsigned mbits = 0u;
    float lmax = NEGBIG;
#pragma unroll
    for (int t = 0; t < 16; ++t) {
        int j  = t * 32 + lane;
        int yt = yrow[j];
        bool msk = (yt != -1);
        int ix = yt < 0 ? 0 : (yt > (N_ - 1) ? (N_ - 1) : yt);
        float s = sanitize_f(srow[ix]);           // L2-resident gather
        float gmv = msk ? s : NEGBIG;
        gm16[t] = gmv;
        mbits |= (unsigned)msk << t;
        lmax = fmaxf(lmax, gmv);
    }
#pragma unroll
    for (int off = 16; off; off >>= 1)
        lmax = fmaxf(lmax, __shfl_xor(lmax, off, 32));
    const float M = lmax;

    // ---- stage e = exp(gm - M) as f16 in LDS (absolute j indexing) ----
#pragma unroll
    for (int t = 0; t < 16; ++t)
        eh[w][t * 32 + lane] = (_Float16)__expf(gm16[t] - M);
    __syncthreads();

    // ---- A = 16x16 upper-triangular ones (U[m][k] = k>=m), K padded to 32 ----
    // 16-bit A 16x32, contiguous k order used here: lane m = lane&15; element e -> K=(lane>>4)*16+e;
    // lanes 16..31 -> K=16..31 (zero padding).
    const int mrow = lane & 15;
    const int hh   = lane >> 4;
    v16h A;
#pragma unroll
    for (int e = 0; e < 16; ++e)
        A[e] = (_Float16)((hh == 0 && e >= mrow) ? 1.0f : 0.0f);

    // ---- B halves: B[u][c'] = e[chunk*16 + u]; lanes 0-15 hold K=0-15 for
    // column N=lane; lanes 16-31 hold K=16-31 which is all zero padding. ----
    v16h B0, B1;
#pragma unroll
    for (int e = 0; e < 16; ++e) { B0[e] = (_Float16)0.0f; B1[e] = (_Float16)0.0f; }
    if (lane < 16) {
#pragma unroll
        for (int e = 0; e < 16; ++e) {
            B0[e] = eh[w][lane * 16 + e];          // chunks 0..15
            B1[e] = eh[w][(lane + 16) * 16 + e];   // chunks 16..31
        }
    }

    v8f d0 = {}; v8f d1 = {};
    d0 = __builtin_amdgcn_wmma_f32_16x16x32_f16(false, A, false, B0, (short)0, d0, false, false);
    d1 = __builtin_amdgcn_wmma_f32_16x16x32_f16(false, A, false, B1, (short)0, d1, false, false);
    // D[t][c] = Sigma_{u>=t} e[c*16+u]; lane holds column c=lane&15, rows t=(lane>>4)*8+r

    // ---- cross-chunk exclusive suffix over chunk totals (D row t=0) ----
    float tl = d0[0], th = d1[0];                       // valid on lanes 0..15
    float va = __shfl(tl, lane & 15, 32);
    float vb = __shfl(th, lane & 15, 32);
    float tot = (lane < 16) ? va : vb;                  // lane holds tot[lane]
    float incl = tot;
#pragma unroll
    for (int off = 1; off < 32; off <<= 1) {
        float v = __shfl_down(incl, off, 32);
        if (lane + off < 32) incl += v;                 // inclusive suffix sum
    }
    float excl  = incl - tot;                           // Sigma_{d > lane} tot[d]
    float exclA = __shfl(excl, lane & 15, 32);          // for chunk  c
    float exclB = __shfl(excl, (lane & 15) + 16, 32);   // for chunk  c+16

    // ---- scatter full suffix sums back to j-order via LDS ----
    const int c0 = lane & 15;
    const int tb = hh * 8;
#pragma unroll
    for (int r = 0; r < 8; ++r) {
        sf[w][c0 * 16 + tb + r]        = d0[r] + exclA;
        sf[w][(c0 + 16) * 16 + tb + r] = d1[r] + exclB;
    }
    __syncthreads();

    // ---- ll = Sigma_{valid j} (g[j] - M - log(suffsum[j])) ----
    float ll = 0.0f;
#pragma unroll
    for (int t = 0; t < 16; ++t) {
        if ((mbits >> t) & 1u) {
            float ss = sf[w][t * 32 + lane];
            ll += gm16[t] - (M + __logf(ss));
        }
    }
    int anyv = (mbits != 0u);
#pragma unroll
    for (int off = 16; off; off >>= 1) {
        ll   += __shfl_xor(ll, off, 32);
        anyv |= __shfl_xor(anyv, off, 32);
    }
    if (lane == 0) {
        row_ll[row]   = anyv ? ll : 0.0f;
        row_used[row] = anyv ? 1.0f : 0.0f;
    }
}

// Deterministic fixed-tree final reduction (no float atomics).
__global__ __launch_bounds__(256) void listnet_reduce_kernel(
    const float* __restrict__ row_ll,
    const float* __restrict__ row_used,
    float* __restrict__ out)
{
    __shared__ float ss[256];
    __shared__ float sn[256];
    const int tid = threadIdx.x;
    float s = 0.0f, n = 0.0f;
    for (int i = tid; i < B_; i += 256) { s += row_ll[i]; n += row_used[i]; }
    ss[tid] = s; sn[tid] = n;
    __syncthreads();
    for (int off = 128; off; off >>= 1) {
        if (tid < off) { ss[tid] += ss[tid + off]; sn[tid] += sn[tid + off]; }
        __syncthreads();
    }
    if (tid == 0) out[0] = (sn[0] > 0.0f) ? (-ss[0] / sn[0]) : 0.0f;
}

extern "C" void kernel_launch(void* const* d_in, const int* in_sizes, int n_in,
                              void* d_out, int out_size, void* d_ws, size_t ws_size,
                              hipStream_t stream) {
    (void)in_sizes; (void)n_in; (void)out_size; (void)ws_size;
    const float* scores = (const float*)d_in[0];   // [B, N] f32
    const int*   ytrue  = (const int*)d_in[1];     // [B, L] int32

    float* row_ll   = (float*)d_ws;                // [B]
    float* row_used = row_ll + B_;                 // [B]

    dim3 grid(B_ / WAVES);
    listnet_rows_kernel<<<grid, BLOCK, 0, stream>>>(scores, ytrue, row_ll, row_used);
    listnet_reduce_kernel<<<1, 256, 0, stream>>>(row_ll, row_used, (float*)d_out);
}
